// MultiHeadAttention_67937792688543
// MI455X (gfx1250) — matrix operand order repaired and checked by reading, not compiled, not run on hardware
//
#include <hip/hip_runtime.h>
#include <hip/hip_bf16.h>

#define HEADS   8
#define DK      64
#define DV      192
#define DMODEL  1536
#define BATCH   4
#define SEQ     1536
#define MTOT    (BATCH*SEQ)     /* 6144 rows for projections */
#define NQK     (HEADS*DK)      /* 512  */
#define NVCOLS  (HEADS*DV)      /* 1536 */

typedef __bf16 bf16;
typedef __attribute__((ext_vector_type(16))) __bf16 bf16x16;
typedef __attribute__((ext_vector_type(8)))  __bf16 bf16x8;
typedef __attribute__((ext_vector_type(8)))  float  f32x8;

#ifndef __has_builtin
#define __has_builtin(x) 0
#endif

#if __has_builtin(__builtin_amdgcn_global_load_async_to_lds_b128)
#define HAVE_ASYNC_LDS 1
typedef int v4i32 __attribute__((vector_size(4 * sizeof(int))));
#else
#define HAVE_ASYNC_LDS 0
#endif

// 16-byte global -> LDS copy. Async DMA path (ASYNCcnt) when available,
// otherwise synchronous load + ds_store (made visible by __syncthreads()).
static __device__ __forceinline__ void cp16_g2l(bf16* l, const bf16* g) {
#if HAVE_ASYNC_LDS
  __builtin_amdgcn_global_load_async_to_lds_b128(
      (__attribute__((address_space(1))) v4i32*)(v4i32*)(void*)g,
      (__attribute__((address_space(3))) v4i32*)(v4i32*)(void*)l, 0, 0);
#else
  *reinterpret_cast<uint4*>(l) = *reinterpret_cast<const uint4*>(g);
#endif
}

static __device__ __forceinline__ void wait_async() {
#if HAVE_ASYNC_LDS
#if __has_builtin(__builtin_amdgcn_s_wait_asynccnt)
  __builtin_amdgcn_s_wait_asynccnt(0);
#else
  asm volatile("s_wait_asynccnt 0x0" ::: "memory");
#endif
#endif
}

static __device__ __forceinline__ f32x8 fzero8() {
  f32x8 z = {0.f,0.f,0.f,0.f,0.f,0.f,0.f,0.f};
  return z;
}

static __device__ __forceinline__ bf16 f2bf(float f) {
  unsigned u = __float_as_uint(f);
  unsigned r = u + 0x7FFFu + ((u >> 16) & 1u);   // round-to-nearest-even
  unsigned short hs = (unsigned short)(r >> 16);
  return __builtin_bit_cast(bf16, hs);
}

static __device__ __forceinline__ bf16x16 join8(bf16x8 lo, bf16x8 hi) {
  return __builtin_shufflevector(lo, hi, 0,1,2,3,4,5,6,7,8,9,10,11,12,13,14,15);
}

// A-fragment 16x32 bf16 (lane m = lane&15, g = lane>>4):
//   elem e in 0..7  -> K = k0 + 8*g + e ; elem e in 8..15 -> K = k0 + 16 + 8*g + (e-8)
static __device__ __forceinline__ bf16x16 load_a(const bf16* rowbase, int k0, int g) {
  bf16x8 lo = *reinterpret_cast<const bf16x8*>(rowbase + k0 + 8*g);
  bf16x8 hi = *reinterpret_cast<const bf16x8*>(rowbase + k0 + 16 + 8*g);
  return join8(lo, hi);
}

// B-fragment 32x16 bf16 (lane n = lane&15, g = lane>>4):
//   elem e in 0..7  -> K = k0 + 8*g + e ; elem e in 8..15 -> K = k0 + 16 + 8*g + (e-8)
static __device__ __forceinline__ bf16x16 load_b(const bf16* colbase, int k0, int g) {
  bf16x8 lo = *reinterpret_cast<const bf16x8*>(colbase + k0 + 8*g);
  bf16x8 hi = *reinterpret_cast<const bf16x8*>(colbase + k0 + 16 + 8*g);
  return join8(lo, hi);
}

static __device__ __forceinline__ f32x8 wmma_bf16(bf16x16 a, bf16x16 b, f32x8 c) {
  return __builtin_amdgcn_wmma_f32_16x16x32_bf16(false, a, false, b, (short)0, c, false, false);
}

// ---------------- fp32 -> bf16 pack (identity layout) ----------------
__global__ void pack_x_kernel(const float* __restrict__ x, bf16* __restrict__ xb, int n) {
  int i = blockIdx.x * blockDim.x + threadIdx.x;
  int stride = gridDim.x * blockDim.x;
  for (; i < n; i += stride) xb[i] = f2bf(x[i]);
}

// ---------------- W [K,N] fp32 -> Wt [N,K] bf16 ----------------
__global__ void pack_wt_kernel(const float* __restrict__ w, bf16* __restrict__ wt,
                               int N, int K) {
  int i = blockIdx.x * blockDim.x + threadIdx.x;
  int tot = N * K;
  int stride = gridDim.x * blockDim.x;
  for (; i < tot; i += stride) {
    int nI = i / K, kI = i - nI * K;
    wt[i] = f2bf(w[(size_t)kI * N + nI]);
  }
}

// ---------------- projection GEMM: C = Xbf @ Wt^T (bf16 WMMA, f32 acc) ----------------
// All 4 waves share the block's 64 output columns -> stage B tile in LDS (double buffered).
// mode 0: Q -> [B,H,S,64] bf16 (scaled)   mode 1: K -> [B,H,S,64] bf16
// mode 2: V -> [B,H,192,S] bf16 (transposed for attention B-fragments)
__global__ __launch_bounds__(128) void proj_gemm_kernel(
    const bf16* __restrict__ A,   // [MTOT, DMODEL]
    const bf16* __restrict__ Bt,  // [N, DMODEL]
    bf16* __restrict__ out, int mode, float scale)
{
  __shared__ __align__(16) bf16 Bl[2][64 * 32];   // [col][k], 4KB per buffer
  const int tid  = threadIdx.x;
  const int w    = tid >> 5;
  const int lane = tid & 31;
  const int n    = lane & 15;
  const int g    = lane >> 4;
  const int rowBase = blockIdx.x * 128 + w * 32;
  const int colBase = blockIdx.y * 64;

  f32x8 acc[2][4];
  #pragma unroll
  for (int mt = 0; mt < 2; mt++)
    #pragma unroll
    for (int nt = 0; nt < 4; nt++) acc[mt][nt] = fzero8();

  const bf16* arow0 = A + (size_t)(rowBase + n) * DMODEL;
  const bf16* arow1 = A + (size_t)(rowBase + 16 + n) * DMODEL;

  // Stage 64 cols x 32 k (4KB): 256 16B chunks, 2 per thread.
  auto stageB = [&](int buf, int k0) {
    #pragma unroll
    for (int c = 0; c < 2; c++) {
      int ch = tid + c * 128;               // 0..255
      int col = ch >> 2, part = ch & 3;
      cp16_g2l(Bl[buf] + ch * 8,
               Bt + (size_t)(colBase + col) * DMODEL + k0 + part * 8);
    }
  };

  stageB(0, 0);
  int it = 0;
  for (int k0 = 0; k0 < DMODEL; k0 += 32, ++it) {
    const int buf = it & 1;
    wait_async();
    __syncthreads();                        // tile ready; prior readers done
    if (k0 + 32 < DMODEL) stageB(buf ^ 1, k0 + 32);

    __builtin_prefetch(arow0 + k0 + 128, 0, 1);   // global_prefetch_b8
    bf16x16 a0 = load_a(arow0, k0, g);
    bf16x16 a1 = load_a(arow1, k0, g);
    #pragma unroll
    for (int nt = 0; nt < 4; nt++) {
      bf16x16 b = load_b(Bl[buf] + (size_t)(nt * 16 + n) * 32, 0, g);
      acc[0][nt] = wmma_bf16(a0, b, acc[0][nt]);
      acc[1][nt] = wmma_bf16(a1, b, acc[1][nt]);
    }
  }

  // C/D layout: lane (n,g), VGPR r -> row = r + 8*g, col = n
  #pragma unroll
  for (int mt = 0; mt < 2; mt++) {
    #pragma unroll
    for (int nt = 0; nt < 4; nt++) {
      #pragma unroll
      for (int r = 0; r < 8; r++) {
        int row = rowBase + mt * 16 + r + 8 * g;    // global token index
        int col = colBase + nt * 16 + n;            // global output column
        int bIdx = row / SEQ, s = row - bIdx * SEQ;
        float v = acc[mt][nt][r] * scale;
        if (mode <= 1) {
          int h = col >> 6, d = col & 63;           // col = h*64 + d
          out[(((size_t)(bIdx * HEADS + h) * SEQ + s) << 6) + d] = f2bf(v);
        } else {
          int h = col / DV, d = col - h * DV;       // col = h*192 + d
          out[((size_t)(bIdx * HEADS + h) * DV + d) * SEQ + s] = f2bf(v);
        }
      }
    }
  }
}

// ---------------- flash attention ----------------
// Block = 4 waves on the SAME (b,h), consecutive query tiles, sharing the
// K (32x64) and V (192x32) tiles of each 32-key step via double-buffered LDS.
__global__ __launch_bounds__(128) void attn_kernel(
    const bf16* __restrict__ Q,   // [B*H, S, 64] pre-scaled bf16
    const bf16* __restrict__ K,   // [B*H, S, 64] bf16
    const bf16* __restrict__ Vt,  // [B*H, 192, S] bf16
    float* __restrict__ out)      // [B, S, H*192] fp32
{
  __shared__ __align__(16) bf16 Kt[2][32 * 64];    // [kpos][dk]  4KB/buf
  __shared__ __align__(16) bf16 Vl[2][192 * 32];   // [dv][kpos] 12KB/buf
  __shared__ __align__(16) bf16 ldsP[4][16 * 32];  // per-wave P staging tile

  const int tid  = threadIdx.x;
  const int w    = tid >> 5;
  const int lane = tid & 31;
  const int n    = lane & 15;
  const int g    = lane >> 4;
  const int QB   = (SEQ / 16) / 4;                 // 24 query-tile groups per bh
  const int bh   = blockIdx.x / QB;
  const int qt   = (blockIdx.x - bh * QB) * 4 + w;
  const int bIdx = bh / HEADS, h = bh - bIdx * HEADS;

  const bf16* Kbase = K  + (size_t)bh * SEQ * DK;
  const bf16* Vbase = Vt + (size_t)bh * DV * SEQ;

  const bf16* qrow = Q + ((size_t)bh * SEQ + qt * 16 + n) * DK;
  bf16x16 qa0 = load_a(qrow, 0, g);
  bf16x16 qa1 = load_a(qrow, 32, g);

  f32x8 o[12];
  #pragma unroll
  for (int t = 0; t < 12; t++) o[t] = fzero8();
  float mrow[8], lrow[8];
  #pragma unroll
  for (int r = 0; r < 8; r++) { mrow[r] = -3.0e38f; lrow[r] = 0.f; }

  bf16* myP = ldsP[w];

  // Stage K tile (4KB = 256 chunks, contiguous in global) and
  // V tile (192 rows x 64B = 768 chunks; LDS side is linear).
  auto stage = [&](int buf, int kp) {
    const bf16* gk = Kbase + (size_t)kp * DK;
    #pragma unroll
    for (int c = 0; c < 2; c++) {
      int ch = tid + c * 128;                       // 0..255
      cp16_g2l(Kt[buf] + ch * 8, gk + ch * 8);
    }
    #pragma unroll
    for (int c = 0; c < 6; c++) {
      int ch = tid + c * 128;                       // 0..767
      int dv = ch >> 2, part = ch & 3;
      cp16_g2l(Vl[buf] + ch * 8, Vbase + (size_t)dv * SEQ + kp + part * 8);
    }
  };

  stage(0, 0);
  int it = 0;
  for (int kp = 0; kp < SEQ; kp += 32, ++it) {
    const int buf = it & 1;
    wait_async();
    __syncthreads();                                // tile ready; prior readers done
    if (kp + 32 < SEQ) stage(buf ^ 1, kp + 32);

    // scores: S = Q (16x64) @ K^T (64x32); K columns come from LDS
    const bf16* kc0 = Kt[buf] + (size_t)n * DK;
    const bf16* kc1 = Kt[buf] + (size_t)(16 + n) * DK;
    f32x8 s0 = fzero8(), s1 = fzero8();
    s0 = wmma_bf16(qa0, load_b(kc0, 0, g), s0);
    s0 = wmma_bf16(qa1, load_b(kc0, 32, g), s0);
    s1 = wmma_bf16(qa0, load_b(kc1, 0, g), s1);
    s1 = wmma_bf16(qa1, load_b(kc1, 32, g), s1);

    // online softmax; row r+8g lives entirely inside this 16-lane half
    #pragma unroll
    for (int r = 0; r < 8; r++) {
      float mx = fmaxf(s0[r], s1[r]);
      mx = fmaxf(mx, __shfl_xor(mx, 1));
      mx = fmaxf(mx, __shfl_xor(mx, 2));
      mx = fmaxf(mx, __shfl_xor(mx, 4));
      mx = fmaxf(mx, __shfl_xor(mx, 8));
      float mnew  = fmaxf(mrow[r], mx);
      float alpha = __expf(mrow[r] - mnew);
      float p0 = __expf(s0[r] - mnew);
      float p1 = __expf(s1[r] - mnew);
      float rs = p0 + p1;
      rs += __shfl_xor(rs, 1);
      rs += __shfl_xor(rs, 2);
      rs += __shfl_xor(rs, 4);
      rs += __shfl_xor(rs, 8);
      lrow[r] = lrow[r] * alpha + rs;
      mrow[r] = mnew;
      #pragma unroll
      for (int t = 0; t < 12; t++) o[t][r] *= alpha;
      // C-layout -> LDS row-major tile [16 rows][32 kpos]
      myP[(r + 8 * g) * 32 + n]      = f2bf(p0);
      myP[(r + 8 * g) * 32 + 16 + n] = f2bf(p1);
    }

    // reload P in A-fragment layout (intra-wave LDS, in-order per DScnt)
    bf16x16 pa = load_a(myP + (size_t)n * 32, 0, g);

    // O += P (16x32) @ V (32x192); V columns come from LDS
    #pragma unroll
    for (int t = 0; t < 12; t++) {
      o[t] = wmma_bf16(pa, load_b(Vl[buf] + (size_t)(t * 16 + n) * 32, 0, g), o[t]);
    }
  }

  // normalize and write fp32 output [B, S, H*192]
  #pragma unroll
  for (int r = 0; r < 8; r++) {
    float inv = 1.0f / lrow[r];
    int q = qt * 16 + r + 8 * g;
    float* orow = out + ((size_t)(bIdx * SEQ + q) * (HEADS * DV)) + h * DV;
    #pragma unroll
    for (int t = 0; t < 12; t++) orow[t * 16 + n] = o[t][r] * inv;
  }
}

extern "C" void kernel_launch(void* const* d_in, const int* in_sizes, int n_in,
                              void* d_out, int out_size, void* d_ws, size_t ws_size,
                              hipStream_t stream) {
  (void)in_sizes; (void)n_in; (void)out_size; (void)ws_size;
  const float* X  = (const float*)d_in[0];
  const float* Wq = (const float*)d_in[1];
  const float* Wk = (const float*)d_in[2];
  const float* Wv = (const float*)d_in[3];
  float* out = (float*)d_out;

  char* p = (char*)d_ws;
  auto take = [&](size_t bytes) {
    char* r = p; p += (bytes + 255) & ~(size_t)255; return r;
  };
  bf16* Xb  = (bf16*)take((size_t)MTOT * DMODEL * 2);
  bf16* Wqt = (bf16*)take((size_t)NQK * DMODEL * 2);
  bf16* Wkt = (bf16*)take((size_t)NQK * DMODEL * 2);
  bf16* Wvt = (bf16*)take((size_t)NVCOLS * DMODEL * 2);
  bf16* Qb  = (bf16*)take((size_t)BATCH * HEADS * SEQ * DK * 2);
  bf16* Kb  = (bf16*)take((size_t)BATCH * HEADS * SEQ * DK * 2);
  bf16* Vtb = (bf16*)take((size_t)BATCH * HEADS * DV * SEQ * 2);

  pack_x_kernel <<<4096, 256, 0, stream>>>(X, Xb, MTOT * DMODEL);
  pack_wt_kernel<<<2048, 256, 0, stream>>>(Wq, Wqt, NQK, DMODEL);
  pack_wt_kernel<<<2048, 256, 0, stream>>>(Wk, Wkt, NQK, DMODEL);
  pack_wt_kernel<<<4096, 256, 0, stream>>>(Wv, Wvt, NVCOLS, DMODEL);

  proj_gemm_kernel<<<dim3(MTOT/128, NQK/64),    128, 0, stream>>>(Xb, Wqt, Qb,  0, 0.125f);
  proj_gemm_kernel<<<dim3(MTOT/128, NQK/64),    128, 0, stream>>>(Xb, Wkt, Kb,  1, 1.0f);
  proj_gemm_kernel<<<dim3(MTOT/128, NVCOLS/64), 128, 0, stream>>>(Xb, Wvt, Vtb, 2, 1.0f);

  attn_kernel<<<(BATCH * HEADS * (SEQ / 16)) / 4, 128, 0, stream>>>(Qb, Kb, Vtb, out);
}
